// GeometryAwarePromptTokenBlock_88656714925352
// MI455X (gfx1250) — matrix operand order repaired and checked by reading, not compiled, not run on hardware
//
#include <hip/hip_runtime.h>
#include <hip/hip_bf16.h>
#include <math.h>

// ---------------------------------------------------------------------------
// GeometryAwarePromptTokenBlock for gfx1250 (MI455X), wave32 + WMMA f16.
// - Token GEMMs: 16x64 strip per wave, A fragment reused across 4 WMMAs.
// - Pointwise-conv GEMMs: 64x64 tile per 4-wave block, B panel staged in LDS
//   via global_load_async_to_lds_b128 (ASYNCcnt), double-buffered.
// ---------------------------------------------------------------------------

typedef __attribute__((ext_vector_type(16))) _Float16 v16h;
typedef __attribute__((ext_vector_type(8)))  _Float16 h8;
typedef __attribute__((ext_vector_type(8)))  float    v8f;

#define CB    4
#define CC    512
#define CH    32
#define CW    32
#define CHW   1024
#define CN    1032      // 8 prompt tokens + 1024 image tokens
#define CNH   8
#define CHD   64
#define CNPR  8
#define CHID  1024
#define CNPAD 1056      // attn/V column padding: multiple of 32
#define CEPS  1e-5f

__device__ __forceinline__ float gelu_f(float x) {
    return 0.5f * x * (1.0f + erff(x * 0.70710678118654752f));
}

// Token coordinates for the geometry bias (matches _coords in reference).
__device__ __forceinline__ void tok_coords(int i, float& cy, float& cx) {
    if (i < CNPR) {                       // 2x4 prompt grid, linspace(0,1,·)
        cy = (float)(i >> 2);             // {0,1}
        cx = (float)(i & 3) * (1.0f / 3.0f);
    } else {
        int j = i - CNPR;                 // 32x32 image grid
        cy = (float)(j >> 5) * (1.0f / 31.0f);
        cx = (float)(j & 31) * (1.0f / 31.0f);
    }
}

struct acc4 { v8f a[4]; };

// ---- CDNA5 async copy: global -> LDS, 16 bytes per lane, ASYNCcnt ----------
__device__ __forceinline__ void async_b128(unsigned ldsByteAddr, const void* g) {
    asm volatile("global_load_async_to_lds_b128 %0, %1, off"
                 :: "v"(ldsByteAddr), "v"((unsigned long long)g)
                 : "memory");
}
__device__ __forceinline__ void wait_async0() {
    asm volatile("s_wait_asynccnt 0x0" ::: "memory");
}
// Generic LDS pointer -> LDS byte offset (ISA 10.2: LDS addr = addr[31:0]).
__device__ __forceinline__ unsigned lds_u32(const void* p) {
    return (unsigned)(unsigned long long)p;
}

// ---------------------------------------------------------------------------
// WMMA 16x64 strip core: D(16x64,f32) = A(16xK,f16) * Bt(64xK,f16)^T
// A row-major over K (lda), Bt row-major over K (ldb): contraction contiguous.
// Row indices clamped to [0,rows) so padded tiles never read OOB.
// A and B both use contiguous K order (k = 16*hi + s), not the ISA table order (wave32).
// ---------------------------------------------------------------------------
__device__ __forceinline__ acc4 wmma_strip_core(
    const _Float16* __restrict__ A, int lda, int Arows,
    const _Float16* __restrict__ Bt, int ldb, int Btrows,
    int K, int m0, int n0)
{
    const int lane = threadIdx.x & 31;
    const int l15  = lane & 15;
    const int hi   = lane >> 4;

    int am = m0 + l15; if (am >= Arows) am = Arows - 1;
    const _Float16* __restrict__ arow = A + (long long)am * lda;

    const _Float16* brow[4];
    #pragma unroll
    for (int j = 0; j < 4; ++j) {
        int bn = n0 + 16 * j + l15;
        if (bn >= Btrows) bn = Btrows - 1;
        brow[j] = Bt + (long long)bn * ldb;
    }

    acc4 acc = {};
    for (int k = 0; k < K; k += 32) {
        // A fragment: 16 contiguous K values starting at k + 16*hi
        h8 alo = *(const h8*)(arow + k + 16 * hi);
        h8 ahi = *(const h8*)(arow + k + 16 * hi + 8);
        v16h a = __builtin_shufflevector(alo, ahi,
                 0,1,2,3,4,5,6,7,8,9,10,11,12,13,14,15);
        #pragma unroll
        for (int j = 0; j < 4; ++j) {
            // B fragment: 16 contiguous K values starting at k + 16*hi
            v16h b = *(const v16h*)(brow[j] + k + 16 * hi);
            acc.a[j] = __builtin_amdgcn_wmma_f32_16x16x32_f16(
                           false, a, false, b, (short)0, acc.a[j], false, false);
        }
    }
    return acc;
}

// ---------------------------------------------------------------------------
// Elementwise / helper kernels
// ---------------------------------------------------------------------------
__global__ void k_f32_to_f16(const float* __restrict__ s,
                             _Float16* __restrict__ d, int n) {
    int i = blockIdx.x * 256 + threadIdx.x;
    if (i < n) d[i] = (_Float16)s[i];
}

__global__ void k_fill_f16_zero(_Float16* __restrict__ p, long long n) {
    long long i = (long long)blockIdx.x * 256 + threadIdx.x;
    if (i < n) p[i] = (_Float16)0.0f;
}

// Depthwise 3x3 conv (SAME) + BN + GELU.  mode 0: src is (B,C,H,W) f32.
// mode 1: gather from jt tokens (rows 8..) shaped (B,N,C).
// Output: f16 (B, HW, C)  -- the transposed GEMM "Bt" operand.
__global__ void k_dwconv_bn_gelu(const float* __restrict__ src,
                                 const float* __restrict__ jt,
                                 const float* __restrict__ w,
                                 const float* __restrict__ bn,
                                 _Float16* __restrict__ dst, int mode)
{
    int idx = blockIdx.x * 256 + threadIdx.x;      // B*C*HW threads
    if (idx >= CB * CC * CHW) return;
    int hw = idx & (CHW - 1);
    int t  = idx >> 10;
    int c  = t & (CC - 1);
    int b  = t >> 9;
    int y = hw >> 5, x = hw & 31;

    float g  = bn[c], bb = bn[CC + c], mm = bn[2 * CC + c], vv = bn[3 * CC + c];
    float sc = g * rsqrtf(vv + CEPS);
    float sh = bb - mm * sc;

    float acc = 0.0f;
    #pragma unroll
    for (int dy = -1; dy <= 1; ++dy) {
        int yy = y + dy;
        if (yy < 0 || yy >= CH) continue;
        #pragma unroll
        for (int dx = -1; dx <= 1; ++dx) {
            int xx = x + dx;
            if (xx < 0 || xx >= CW) continue;
            float v;
            if (mode == 0)
                v = src[(((long long)b * CC + c) << 10) + yy * CW + xx];
            else
                v = jt[((long long)(b * CN + CNPR + yy * CW + xx)) * CC + c];
            acc += v * w[c * 9 + (dy + 1) * 3 + (dx + 1)];
        }
    }
    float val = gelu_f(acc * sc + sh);
    dst[((long long)(b * CHW + hw)) * CC + c] = (_Float16)val;
}

// fused (B,C,H,W) -> jt image token rows (B, 8+hw, C)
__global__ void k_jt_img(const float* __restrict__ fused,
                         float* __restrict__ jt)
{
    int idx = blockIdx.x * 256 + threadIdx.x;
    if (idx >= CB * CC * CHW) return;
    int hw = idx & (CHW - 1);
    int t  = idx >> 10;
    int c  = t & (CC - 1);
    int b  = t >> 9;
    jt[((long long)(b * CN + CNPR + hw)) * CC + c] =
        fused[(((long long)b * CC + c) << 10) + hw];
}

// Pool pp (B,C,HW) over 2x4 grid, add prompt_embed, LayerNorm over C,
// write jt rows 0..7.  One block per (b,p), 256 threads, 2 channels each.
__global__ void k_pool_prompt(const float* __restrict__ pp,
                              const float* __restrict__ embed,
                              const float* __restrict__ ln,
                              float* __restrict__ jt)
{
    __shared__ float red[256];
    int bp = blockIdx.x;            // 0..31
    int b = bp >> 3, p = bp & 7;
    int py = p >> 2, px = p & 3;
    int t = threadIdx.x;

    float v[2];
    #pragma unroll
    for (int s = 0; s < 2; ++s) {
        int c = t + s * 256;
        const float* base = pp + (((long long)b * CC + c) << 10);
        float sum = 0.0f;
        for (int y = py * 16; y < py * 16 + 16; ++y)
            for (int x = px * 8; x < px * 8 + 8; ++x)
                sum += base[y * CW + x];
        v[s] = sum * (1.0f / 128.0f) + embed[p * CC + c];
    }
    red[t] = v[0] + v[1];
    __syncthreads();
    for (int s = 128; s > 0; s >>= 1) {
        if (t < s) red[t] += red[t + s];
        __syncthreads();
    }
    float mu = red[0] * (1.0f / CC);
    __syncthreads();
    red[t] = v[0] * v[0] + v[1] * v[1];
    __syncthreads();
    for (int s = 128; s > 0; s >>= 1) {
        if (t < s) red[t] += red[t + s];
        __syncthreads();
    }
    float var = red[0] * (1.0f / CC) - mu * mu;
    float rs = rsqrtf(var + CEPS);
    #pragma unroll
    for (int s = 0; s < 2; ++s) {
        int c = t + s * 256;
        jt[((long long)(b * CN + p)) * CC + c] =
            (v[s] - mu) * rs * ln[c] + ln[CC + c];
    }
}

// LayerNorm of jt rows (B*N, C) -> f16 dst.
__global__ void k_ln_rows(const float* __restrict__ src,
                          const float* __restrict__ ln,
                          _Float16* __restrict__ dst)
{
    __shared__ float red[256];
    long long r = blockIdx.x;
    const float* x = src + r * CC;
    int t = threadIdx.x;
    float v0 = x[t], v1 = x[t + 256];
    red[t] = v0 + v1;
    __syncthreads();
    for (int s = 128; s > 0; s >>= 1) {
        if (t < s) red[t] += red[t + s];
        __syncthreads();
    }
    float mu = red[0] * (1.0f / CC);
    __syncthreads();
    red[t] = v0 * v0 + v1 * v1;
    __syncthreads();
    for (int s = 128; s > 0; s >>= 1) {
        if (t < s) red[t] += red[t + s];
        __syncthreads();
    }
    float var = red[0] * (1.0f / CC) - mu * mu;
    float rs = rsqrtf(var + CEPS);
    dst[r * CC + t]       = (_Float16)((v0 - mu) * rs * ln[t] + ln[CC + t]);
    dst[r * CC + t + 256] = (_Float16)((v1 - mu) * rs * ln[t + 256] + ln[CC + t + 256]);
}

// ds reduction: jd[b, 8+hw] = sum_c gp_pw_w[c]*dw[b,hw,c] + gp_pw_b.
// One wave per (b,hw).
__global__ void k_ds(const _Float16* __restrict__ dw,
                     const float* __restrict__ pw,
                     const float* __restrict__ pwb,
                     float* __restrict__ jd)
{
    int wid  = (blockIdx.x * 256 + threadIdx.x) >> 5;
    int lane = threadIdx.x & 31;
    if (wid >= CB * CHW) return;
    int b = wid >> 10, hw = wid & (CHW - 1);
    const _Float16* row = dw + ((long long)(b * CHW + hw)) * CC;
    float acc = 0.0f;
    for (int c = lane; c < CC; c += 32) acc += (float)row[c] * pw[c];
    for (int o = 16; o > 0; o >>= 1) acc += __shfl_xor(acc, o, 32);
    if (lane == 0) jd[b * CN + CNPR + hw] = acc + pwb[0];
}

// jd[b, p] = average of jd image part over the 2x4 pool cell.
__global__ void k_jd_pool(float* __restrict__ jd)
{
    int t = threadIdx.x;
    if (t >= CB * CNPR) return;
    int b = t >> 3, p = t & 7;
    int py = p >> 2, px = p & 3;
    float sum = 0.0f;
    for (int y = py * 16; y < py * 16 + 16; ++y)
        for (int x = px * 8; x < px * 8 + 8; ++x)
            sum += jd[b * CN + CNPR + y * CW + x];
    jd[b * CN + p] = sum * (1.0f / 128.0f);
}

// Row softmax, in place on f16 attn rows (length CN, row stride CNPAD).
__global__ void k_softmax(_Float16* __restrict__ attn)
{
    __shared__ float red[256];
    long long row = blockIdx.x;                    // B*NH*N rows
    _Float16* p = attn + row * CNPAD;
    int t = threadIdx.x;
    float v[5];
    int cnt = 0;
    float lmax = -1e30f;
    for (int i = t; i < CN; i += 256) {
        float x = (float)p[i];
        v[cnt++] = x;
        if (x > lmax) lmax = x;
    }
    red[t] = lmax;
    __syncthreads();
    for (int s = 128; s > 0; s >>= 1) {
        if (t < s) red[t] = fmaxf(red[t], red[t + s]);
        __syncthreads();
    }
    float gmax = red[0];
    __syncthreads();
    float lsum = 0.0f;
    for (int j = 0; j < cnt; ++j) {
        float e = expf(v[j] - gmax);
        v[j] = e;
        lsum += e;
    }
    red[t] = lsum;
    __syncthreads();
    for (int s = 128; s > 0; s >>= 1) {
        if (t < s) red[t] += red[t + s];
        __syncthreads();
    }
    float inv = 1.0f / red[0];
    cnt = 0;
    for (int i = t; i < CN; i += 256) p[i] = (_Float16)(v[cnt++] * inv);
}

// ---------------------------------------------------------------------------
// Pointwise conv GEMM with LDS-staged B panel (async copy, double buffered).
// Block = 128 threads (4 waves) -> 64x64 output tile; wave w owns rows
// m0 = blockIdx.y*64 + 16w.  B panel per K-step: 64 rows x 32 halves = 4 KB.
// ---------------------------------------------------------------------------
__global__ void __launch_bounds__(128)
k_gemm_pw(const _Float16* __restrict__ W,
          const _Float16* __restrict__ BtBase,
          float* __restrict__ outF,
          const float* __restrict__ bn,
          const float* __restrict__ addSrc,
          int geluFlag)
{
    __shared__ _Float16 ldsB[2][64 * 32];

    int z  = blockIdx.z;                  // batch
    int tid = threadIdx.x;
    int m0 = blockIdx.y * 64 + ((tid >> 5) << 4);
    int n0 = blockIdx.x * 64;
    const _Float16* Bt = BtBase + (long long)z * CHW * CC;

    // async loader: thread t moves 32 B of row (t>>1), half-chunk (t&1)
    int lrow = tid >> 1;                  // 0..63
    int lch  = (tid & 1) * 16;            // halves within the 32-wide panel
    const _Float16* gsrc = Bt + (long long)(n0 + lrow) * CC + lch;
    unsigned d0 = lds_u32(&ldsB[0][lrow * 32 + lch]);
    unsigned d1 = lds_u32(&ldsB[1][lrow * 32 + lch]);

    int lane = tid & 31, l15 = lane & 15, hi = lane >> 4;
    const _Float16* arow = W + (long long)(m0 + l15) * CC;

    // preload K-step 0 into buffer 0
    async_b128(d0,      gsrc);
    async_b128(d0 + 16, gsrc + 8);
    wait_async0();
    __syncthreads();

    acc4 acc = {};
    for (int k = 0; k < CC; k += 32) {
        int buf = (k >> 5) & 1;
        // prefetch next K-step into the other buffer
        if (k + 32 < CC) {
            unsigned nd = buf ? d0 : d1;
            const _Float16* gn = gsrc + k + 32;
            async_b128(nd,      gn);
            async_b128(nd + 16, gn + 8);
        }
        h8 alo = *(const h8*)(arow + k + 16 * hi);
        h8 ahi = *(const h8*)(arow + k + 16 * hi + 8);
        v16h a = __builtin_shufflevector(alo, ahi,
                 0,1,2,3,4,5,6,7,8,9,10,11,12,13,14,15);
        const _Float16* bbase = &ldsB[buf][0];
        #pragma unroll
        for (int j = 0; j < 4; ++j) {
            v16h b = *(const v16h*)(bbase + (16 * j + l15) * 32 + 16 * hi);
            acc.a[j] = __builtin_amdgcn_wmma_f32_16x16x32_f16(
                           false, a, false, b, (short)0, acc.a[j], false, false);
        }
        wait_async0();
        __syncthreads();
    }

    #pragma unroll
    for (int r = 0; r < 8; ++r) {
        int m = m0 + r + 8 * hi;          // output channel
        float g = bn[m], bb = bn[CC + m], mm = bn[2 * CC + m], vv = bn[3 * CC + m];
        float sc = g * rsqrtf(vv + CEPS);
        float sh = bb - mm * sc;
        #pragma unroll
        for (int j = 0; j < 4; ++j) {
            int n = n0 + 16 * j + l15;
            float val = acc.a[j][r] * sc + sh;
            if (geluFlag) val = gelu_f(val);
            long long o = (((long long)z * CC + m) << 10) + n;
            if (addSrc) val += addSrc[o];
            outF[o] = val;
        }
    }
}

// Tokens x weight -> f16 output.  transOut=0: row-major [m][n]; =1: [n][m]
// (used to build V^T).  Optional per-column bias + GELU (MLP1).
__global__ void k_gemm_tokw_f16(const _Float16* __restrict__ A, int lda, int Arows,
                                const _Float16* __restrict__ Bt, int ldb, int Btrows,
                                int K, _Float16* __restrict__ outH, int ldo,
                                int zDiv,
                                long long aZa, long long aZb,
                                long long bZa, long long bZb,
                                long long oZa, long long oZb,
                                const float* __restrict__ bias,
                                int geluFlag, int transOut,
                                int Mlimit, int Nlimit)
{
    int z = blockIdx.z;
    int zq = z / zDiv, zr = z - zq * zDiv;
    long long aoff = (long long)zq * aZa + (long long)zr * aZb;
    long long boff = (long long)zq * bZa + (long long)zr * bZb;
    long long ooff = (long long)zq * oZa + (long long)zr * oZb;
    int m0 = blockIdx.y * 16;
    int n0 = blockIdx.x * 64;

    acc4 acc = wmma_strip_core(A + aoff, lda, Arows, Bt + boff, ldb, Btrows,
                               K, m0, n0);

    int lane = threadIdx.x & 31, l15 = lane & 15, hi = lane >> 4;
    if (!transOut) {
        #pragma unroll
        for (int r = 0; r < 8; ++r) {
            int m = m0 + r + 8 * hi;
            if (m < Mlimit) {
                #pragma unroll
                for (int j = 0; j < 4; ++j) {
                    int n = n0 + 16 * j + l15;
                    if (n < Nlimit) {
                        float v = acc.a[j][r];
                        if (bias) v += bias[n];
                        if (geluFlag) v = gelu_f(v);
                        outH[ooff + (long long)m * ldo + n] = (_Float16)v;
                    }
                }
            }
        }
    } else {
        #pragma unroll
        for (int j = 0; j < 4; ++j) {
            int n = n0 + 16 * j + l15;
            if (n >= Nlimit) continue;
            #pragma unroll
            for (int r = 0; r < 8; ++r) {
                int m = m0 + 8 * hi + r;
                if (m < Mlimit)
                    outH[ooff + (long long)n * ldo + m] = (_Float16)acc.a[j][r];
            }
        }
    }
}

// Tokens x weight -> accumulate into f32 (residual adds), optional bias.
__global__ void k_gemm_tokw_accf32(const _Float16* __restrict__ A, int lda, int Arows,
                                   const _Float16* __restrict__ Bt, int ldb, int Btrows,
                                   int K, float* __restrict__ outF, int ldo,
                                   long long aZ, long long oZ,
                                   const float* __restrict__ bias,
                                   int Mlimit, int Nlimit)
{
    int z = blockIdx.z;
    int m0 = blockIdx.y * 16;
    int n0 = blockIdx.x * 64;
    acc4 acc = wmma_strip_core(A + (long long)z * aZ, lda, Arows,
                               Bt, ldb, Btrows, K, m0, n0);
    int lane = threadIdx.x & 31, l15 = lane & 15, hi = lane >> 4;
    long long ooff = (long long)z * oZ;
    #pragma unroll
    for (int r = 0; r < 8; ++r) {
        int m = m0 + r + 8 * hi;
        if (m < Mlimit) {
            #pragma unroll
            for (int j = 0; j < 4; ++j) {
                int n = n0 + 16 * j + l15;
                if (n < Nlimit) {
                    float v = acc.a[j][r];
                    if (bias) v += bias[n];
                    outF[ooff + (long long)m * ldo + n] += v;
                }
            }
        }
    }
}

// Attention logits: per (b,h): q.k^T * HD^-0.5 + geometry bias -> f16 attn.
__global__ void k_gemm_logits(const _Float16* __restrict__ qh,
                              const _Float16* __restrict__ kh,
                              _Float16* __restrict__ attn,
                              const float* __restrict__ jd,
                              const float* __restrict__ gpw)
{
    int z = blockIdx.z;                   // b*NH + h
    int b = z >> 3, h = z & 7;
    long long qoff = (long long)b * CN * CC + h * CHD;
    int m0 = blockIdx.y * 16;
    int n0 = blockIdx.x * 64;

    acc4 acc = wmma_strip_core(qh + qoff, CC, CN, kh + qoff, CC, CN,
                               CHD, m0, n0);

    float decay = logf(1.0f - exp2f(-1.0f - 0.5f * (float)h));
    float w0 = gpw[0], w1 = gpw[1];

    int lane = threadIdx.x & 31, l15 = lane & 15, hi = lane >> 4;
    float cyn[4], cxn[4], jdn[4];
    #pragma unroll
    for (int j = 0; j < 4; ++j) {
        int n = n0 + 16 * j + l15;
        int nc = n < CN ? n : CN - 1;
        tok_coords(nc, cyn[j], cxn[j]);
        jdn[j] = jd[b * CN + nc];
    }
    long long zoff = (long long)z * CN * CNPAD;
    #pragma unroll
    for (int r = 0; r < 8; ++r) {
        int m = m0 + r + 8 * hi;
        if (m < CN) {
            float cym, cxm;
            tok_coords(m, cym, cxm);
            float jdm = jd[b * CN + m];
            #pragma unroll
            for (int j = 0; j < 4; ++j) {
                int n = n0 + 16 * j + l15;
                if (n < CN) {
                    float pos = fabsf(cym - cyn[j]) + fabsf(cxm - cxn[j]);
                    float dbias = decay * (w0 * pos + w1 * fabsf(jdm - jdn[j]));
                    float v = acc.a[j][r] * 0.125f + dbias;   // HD^-0.5 = 1/8
                    attn[zoff + (long long)m * CNPAD + n] = (_Float16)v;
                }
            }
        }
    }
}

// ---------------------------------------------------------------------------
// Host-side orchestration
// ---------------------------------------------------------------------------
extern "C" void kernel_launch(void* const* d_in, const int* in_sizes, int n_in,
                              void* d_out, int out_size, void* d_ws, size_t ws_size,
                              hipStream_t stream)
{
    (void)in_sizes; (void)n_in; (void)out_size;

    const float* fused     = (const float*)d_in[0];
    const float* depth     = (const float*)d_in[1];
    const float* pg_dw_w   = (const float*)d_in[2];
    const float* pg_bn1    = (const float*)d_in[3];
    const float* pg_pw_w   = (const float*)d_in[4];
    const float* pg_bn2    = (const float*)d_in[5];
    const float* pr_embed  = (const float*)d_in[6];
    const float* pg_ln     = (const float*)d_in[7];
    const float* gp_dw_w   = (const float*)d_in[8];
    const float* gp_bn     = (const float*)d_in[9];
    const float* gp_pw_w   = (const float*)d_in[10];
    const float* gp_pw_b   = (const float*)d_in[11];
    const float* gp_weight = (const float*)d_in[12];
    const float* ln1       = (const float*)d_in[13];
    const float* wq        = (const float*)d_in[14];
    const float* wk        = (const float*)d_in[15];
    const float* wv        = (const float*)d_in[16];
    const float* wo        = (const float*)d_in[17];
    const float* ln2       = (const float*)d_in[18];
    const float* mlp_w1    = (const float*)d_in[19];
    const float* mlp_b1    = (const float*)d_in[20];
    const float* mlp_w2    = (const float*)d_in[21];
    const float* mlp_b2    = (const float*)d_in[22];
    const float* op_dw_w   = (const float*)d_in[23];
    const float* op_bn1    = (const float*)d_in[24];
    const float* op_pw_w   = (const float*)d_in[25];
    const float* op_bn2    = (const float*)d_in[26];
    float* outp = (float*)d_out;

    // ---- workspace carve-up (256B aligned) ----
    size_t off = 0;
    char* base = (char*)d_ws;
    auto take = [&](size_t bytes) -> char* {
        char* p = base + off;
        off += (bytes + 255) & ~(size_t)255;
        return p;
    };
    _Float16* bufDW = (_Float16*)take((size_t)CB * CHW * CC * 2);   // dwconv out (B,HW,C)
    float*    pp    = (float*)   take((size_t)CB * CC * CHW * 4);   // pg pw out (B,C,HW)
    float*    jt    = (float*)   take((size_t)CB * CN * CC * 4);    // joint tokens
    _Float16* xh    = (_Float16*)take((size_t)CB * CN * CC * 2);    // LN out
    _Float16* qh    = (_Float16*)take((size_t)CB * CN * CC * 2);
    _Float16* kh    = (_Float16*)take((size_t)CB * CN * CC * 2);
    _Float16* vth   = (_Float16*)take((size_t)CB * CC * CNPAD * 2); // V^T (B,NH,HD,NPAD)
    _Float16* aoh   = (_Float16*)take((size_t)CB * CN * CC * 2);
    _Float16* hbuf  = (_Float16*)take((size_t)CB * CN * CHID * 2);
    _Float16* attn  = (_Float16*)take((size_t)CB * CNH * CN * CNPAD * 2);
    float*    jd    = (float*)   take((size_t)CB * CN * 4);
    _Float16* pgpwh = (_Float16*)take((size_t)CC * CC * 2);
    _Float16* wqh   = (_Float16*)take((size_t)CC * CC * 2);
    _Float16* wkh   = (_Float16*)take((size_t)CC * CC * 2);
    _Float16* wvh   = (_Float16*)take((size_t)CC * CC * 2);
    _Float16* woh   = (_Float16*)take((size_t)CC * CC * 2);
    _Float16* w1h   = (_Float16*)take((size_t)CHID * CC * 2);
    _Float16* w2h   = (_Float16*)take((size_t)CC * CHID * 2);
    _Float16* oppwh = (_Float16*)take((size_t)CC * CC * 2);
    if (off > ws_size) return;   // workspace too small: deterministic no-op

    const long long NC = (long long)CN * CC;

    // ---- weight conversions to f16 ----
    {
        struct CV { const float* s; _Float16* d; int n; } cv[] = {
            {pg_pw_w, pgpwh, CC * CC}, {wq, wqh, CC * CC}, {wk, wkh, CC * CC},
            {wv, wvh, CC * CC}, {wo, woh, CC * CC}, {mlp_w1, w1h, CHID * CC},
            {mlp_w2, w2h, CC * CHID}, {op_pw_w, oppwh, CC * CC}};
        for (auto& c : cv)
            k_f32_to_f16<<<(c.n + 255) / 256, 256, 0, stream>>>(c.s, c.d, c.n);
    }

    const int EW = CB * CC * CHW;   // elementwise grid size

    // ---- DepthPromptGenerator ----
    k_dwconv_bn_gelu<<<(EW + 255) / 256, 256, 0, stream>>>(
        depth, nullptr, pg_dw_w, pg_bn1, bufDW, 0);
    k_gemm_pw<<<dim3(CHW / 64, CC / 64, CB), 128, 0, stream>>>(
        pgpwh, bufDW, pp, pg_bn2, nullptr, 1);
    k_pool_prompt<<<CB * CNPR, 256, 0, stream>>>(pp, pr_embed, pg_ln, jt);
    k_jt_img<<<(EW + 255) / 256, 256, 0, stream>>>(fused, jt);

    // ---- GeometryPriorGenerator (jd) ----
    k_dwconv_bn_gelu<<<(EW + 255) / 256, 256, 0, stream>>>(
        depth, nullptr, gp_dw_w, gp_bn, bufDW, 0);
    k_ds<<<(CB * CHW * 32) / 256, 256, 0, stream>>>(bufDW, gp_pw_w, gp_pw_b, jd);
    k_jd_pool<<<1, 32, 0, stream>>>(jd);

    // ---- Mixer: LN1, QKV ----
    k_ln_rows<<<CB * CN, 256, 0, stream>>>(jt, ln1, xh);
    const int MT  = (CN + 15) / 16;    // 65 token tiles (rows)
    const int NT4 = (CN + 63) / 64;    // 17 token strips (cols)
    k_gemm_tokw_f16<<<dim3(CC / 64, MT, CB), 32, 0, stream>>>(
        xh, CC, CN, wqh, CC, CC, CC, qh, CC,
        1, NC, 0, 0, 0, NC, 0, nullptr, 0, 0, CN, CC);
    k_gemm_tokw_f16<<<dim3(CC / 64, MT, CB), 32, 0, stream>>>(
        xh, CC, CN, wkh, CC, CC, CC, kh, CC,
        1, NC, 0, 0, 0, NC, 0, nullptr, 0, 0, CN, CC);
    k_fill_f16_zero<<<(unsigned)(((long long)CB * CC * CNPAD + 255) / 256), 256, 0, stream>>>(
        vth, (long long)CB * CC * CNPAD);
    k_gemm_tokw_f16<<<dim3(CC / 64, MT, CB), 32, 0, stream>>>(
        xh, CC, CN, wvh, CC, CC, CC, vth, CNPAD,
        1, NC, 0, 0, 0, (long long)CC * CNPAD, 0, nullptr, 0, 1, CN, CC);

    // ---- logits + softmax + attn*V ----
    k_fill_f16_zero<<<(unsigned)(((long long)CB * CNH * CN * CNPAD + 255) / 256), 256, 0, stream>>>(
        attn, (long long)CB * CNH * CN * CNPAD);
    k_gemm_logits<<<dim3(NT4, MT, CB * CNH), 32, 0, stream>>>(
        qh, kh, attn, jd, gp_weight);
    k_softmax<<<CB * CNH * CN, 256, 0, stream>>>(attn);
    k_gemm_tokw_f16<<<dim3(CHD / 64, MT, CB * CNH), 32, 0, stream>>>(
        attn, CNPAD, CN, vth, CNPAD, CHD, CNPAD, aoh, CC,
        CNH,
        (long long)CNH * CN * CNPAD, (long long)CN * CNPAD,
        (long long)CC * CNPAD, (long long)CHD * CNPAD,
        NC, CHD,
        nullptr, 0, 0, CN, CHD);

    // ---- output projection residual ----
    k_gemm_tokw_accf32<<<dim3(CC / 64, MT, CB), 32, 0, stream>>>(
        aoh, CC, CN, woh, CC, CC, CC, jt, CC, NC, NC, nullptr, CN, CC);

    // ---- MLP ----
    k_ln_rows<<<CB * CN, 256, 0, stream>>>(jt, ln2, xh);
    k_gemm_tokw_f16<<<dim3(CHID / 64, MT, CB), 32, 0, stream>>>(
        xh, CC, CN, w1h, CC, CHID, CC, hbuf, CHID,
        1, NC, 0, 0, 0, (long long)CN * CHID, 0, mlp_b1, 1, 0, CN, CHID);
    k_gemm_tokw_accf32<<<dim3(CC / 64, MT, CB), 32, 0, stream>>>(
        hbuf, CHID, CN, w2h, CHID, CC, CHID, jt, CC,
        (long long)CN * CHID, NC, mlp_b2, CN, CC);

    // ---- out_proj: dwconv(enh)+bn+gelu, pointwise+bn2, + fused ----
    k_dwconv_bn_gelu<<<(EW + 255) / 256, 256, 0, stream>>>(
        nullptr, jt, op_dw_w, op_bn1, bufDW, 1);
    k_gemm_pw<<<dim3(CHW / 64, CC / 64, CB), 128, 0, stream>>>(
        oppwh, bufDW, outp, op_bn2, fused, 0);
}
